// SpectralGradients_51805895524429
// MI455X (gfx1250) — compile-verified, matrix operand order checked by reading, not run on hardware
//
#include <hip/hip_runtime.h>
#include <hip/hip_bf16.h>
#include <math.h>

typedef __attribute__((ext_vector_type(16))) _Float16     v16h;
typedef __attribute__((ext_vector_type(8)))  float        v8f;
typedef __attribute__((ext_vector_type(4)))  unsigned int v4u;
typedef __attribute__((ext_vector_type(8)))  int          v8i;
typedef __attribute__((ext_vector_type(4)))  int          v4i;

#define TT   3840
#define BBAT 32
#define HH   256
#define MMO  5
#define FF   64
#define PAD  18
#define TEXT (TT + 2*PAD)      /* 3876 */
#define NROW ((FF+1)*BBAT)     /* 2080 rows of V: [x ; ends0..ends63] */
#define NFB  (FF*BBAT)         /* 2048 (filter,batch) pairs */

/* ---- workspace layout (float units); total ~21.2M floats (~85 MB) ---- */
#define OFF_COEF 0                                 /* 64*20               */
#define OFF_V    2048                              /* NROW*TT             */
#define OFF_TMP  (OFF_V + NROW*TT)                 /* NFB*TEXT            */
#define OFF_HPRE (OFF_TMP + NFB*TEXT)              /* NROW*HH             */
#define OFF_FN   (OFF_HPRE + NROW*HH)              /* NROW*MMO            */
#define OFF_U    (OFF_FN + NROW*MMO + 64)          /* NFB*8*32*16 halves  */
#define OFF_W1H  (OFF_U + (NFB*8*32*16)/2)         /* TT*HH halves        */

/* K-offset inside a 16x32 (or 32x16) f16 WMMA fragment:
   lanes 0-15 (hi=0): elems 0..7 -> K 0..7,  elems 8..15 -> K 16..23
   lanes16-31 (hi=1): elems 0..7 -> K 8..15, elems 8..15 -> K 24..31   */
__device__ __forceinline__ int kof(int e, int hi) {
  return (e < 8) ? (e + (hi ? 8 : 0)) : (e + (hi ? 16 : 8));
}

/* ================= 1) Butterworth design (order 5) ================= */
__global__ void k_design(float* coef) {
  int f = threadIdx.x;
  if (f >= FF) return;
  float* C = coef + f * 20;
  double cut = 63.0 - (double)f;
  if (cut <= 0.0) {
    for (int i = 0; i < 20; ++i) C[i] = 0.f;
    C[17] = 1.f;   /* zero-filter flag */
    return;
  }
  double Wn = cut / 64.0;
  double warped = 4.0 * tan(M_PI * Wn * 0.5);
  double pr[5], pim[5];
  double kgain = 1.0;
  for (int j = 0; j < 5; ++j) {
    double mm = -4.0 + 2.0 * j;            /* -4,-2,0,2,4 */
    double ang = M_PI * mm / 10.0;
    pr[j]  = -cos(ang) * warped;
    pim[j] = -sin(ang) * warped;
    kgain *= warped;
  }
  /* kd = k * Re(1/prod(4-p)) */
  double qr = 1.0, qi = 0.0;
  for (int j = 0; j < 5; ++j) {
    double ar = 4.0 - pr[j], ai = -pim[j];
    double nr = qr * ar - qi * ai, ni = qr * ai + qi * ar;
    qr = nr; qi = ni;
  }
  double kd = kgain * (qr / (qr * qr + qi * qi));
  /* pd = (4+p)/(4-p) (bilinear) */
  double dr[5], di[5];
  for (int j = 0; j < 5; ++j) {
    double nr = 4.0 + pr[j], ni = pim[j];
    double er = 4.0 - pr[j], ei = -pim[j];
    double den = er * er + ei * ei;
    dr[j] = (nr * er + ni * ei) / den;
    di[j] = (ni * er - nr * ei) / den;
  }
  /* a = real(poly(pd)); poly via conv with [1,-r] */
  double cr[6] = {1, 0, 0, 0, 0, 0}, ci[6] = {0, 0, 0, 0, 0, 0};
  for (int j = 0; j < 5; ++j) {
    for (int i = j + 1; i >= 1; --i) {
      double nr = cr[i] - (dr[j] * cr[i - 1] - di[j] * ci[i - 1]);
      double ni = ci[i] - (dr[j] * ci[i - 1] + di[j] * cr[i - 1]);
      cr[i] = nr; ci[i] = ni;
    }
  }
  const double binom[6] = {1, 5, 10, 10, 5, 1};
  double bco[6], aco[6];
  for (int i = 0; i < 6; ++i) { bco[i] = kd * binom[i]; aco[i] = cr[i]; }
  /* lfilter_zi: solve (I - comp^T) zi = b[1:] - a[1:]*b0 */
  double Mz[5][5], Bv[5], zi[5];
  for (int i = 0; i < 5; ++i) {
    for (int j = 0; j < 5; ++j) {
      double compT = (j == 0) ? -aco[i + 1] : ((i == j - 1) ? 1.0 : 0.0);
      Mz[i][j] = ((i == j) ? 1.0 : 0.0) - compT;
    }
    Bv[i] = bco[i + 1] - aco[i + 1] * bco[0];
  }
  for (int c0 = 0; c0 < 5; ++c0) {
    int piv = c0; double best = fabs(Mz[c0][c0]);
    for (int r = c0 + 1; r < 5; ++r)
      if (fabs(Mz[r][c0]) > best) { best = fabs(Mz[r][c0]); piv = r; }
    if (piv != c0) {
      for (int cc = 0; cc < 5; ++cc) { double t = Mz[c0][cc]; Mz[c0][cc] = Mz[piv][cc]; Mz[piv][cc] = t; }
      double t = Bv[c0]; Bv[c0] = Bv[piv]; Bv[piv] = t;
    }
    double inv = 1.0 / Mz[c0][c0];
    for (int r = c0 + 1; r < 5; ++r) {
      double fct = Mz[r][c0] * inv;
      for (int cc = c0; cc < 5; ++cc) Mz[r][cc] -= fct * Mz[c0][cc];
      Bv[r] -= fct * Bv[c0];
    }
  }
  for (int i = 4; i >= 0; --i) {
    double s = Bv[i];
    for (int j = i + 1; j < 5; ++j) s -= Mz[i][j] * zi[j];
    zi[i] = s / Mz[i][i];
  }
  for (int i = 0; i < 6; ++i) C[i] = (float)bco[i];
  for (int i = 0; i < 6; ++i) C[6 + i] = (float)aco[i];
  for (int i = 0; i < 5; ++i) C[12 + i] = (float)zi[i];
  C[17] = 0.f;
}

/* ================= 2) V[0..31] = x ================= */
__global__ void k_copy_x(const float* __restrict__ x, float* __restrict__ V) {
  int i = blockIdx.x * blockDim.x + threadIdx.x;
  if (i < BBAT * TT) V[i] = x[i];
}

/* ================= 2b) W1h = (f16)W1, one-time convert ================= */
__global__ void k_cvt(const float* __restrict__ W1, _Float16* __restrict__ W1h) {
  int i = blockIdx.x * blockDim.x + threadIdx.x;
  if (i < TT * HH) W1h[i] = (_Float16)W1[i];
}

/* ================= 3) filtfilt per (f,b) ================= */
__global__ void k_filt(const float* __restrict__ x, const float* __restrict__ coef,
                       float* __restrict__ V, float* __restrict__ tmp) {
  int p = blockIdx.x * blockDim.x + threadIdx.x;
  if (p >= NFB) return;
  int f = p >> 5, b = p & 31;
  const float* C = coef + f * 20;
  float* vout = V + (size_t)(p + BBAT) * TT;   /* ends[f] -> V[f+1] */
  if (C[17] != 0.f) {
    for (int t = 0; t < TT; ++t) vout[t] = 0.f;
    return;
  }
  float bb[6], aa[6], zv[5];
  for (int i = 0; i < 6; ++i) { bb[i] = C[i]; aa[i] = C[6 + i]; }
  for (int i = 0; i < 5; ++i) zv[i] = C[12 + i];
  const float* xr = x + (size_t)b * TT;
  float* tr = tmp + (size_t)p * TEXT;
  /* forward pass over reflected-extension */
  float x0e = 2.f * xr[0] - xr[PAD];
  float z[5];
  for (int i = 0; i < 5; ++i) z[i] = zv[i] * x0e;
  for (int j = 0; j < TEXT; ++j) {
    float xt;
    if (j < PAD)            xt = 2.f * xr[0] - xr[PAD - j];
    else if (j < PAD + TT)  xt = xr[j - PAD];
    else { int jr = j - PAD - TT; xt = 2.f * xr[TT - 1] - xr[TT - 2 - jr]; }
    float y = bb[0] * xt + z[0];
    z[0] = xt * bb[1] + z[1] - y * aa[1];
    z[1] = xt * bb[2] + z[2] - y * aa[2];
    z[2] = xt * bb[3] + z[3] - y * aa[3];
    z[3] = xt * bb[4] + z[4] - y * aa[4];
    z[4] = xt * bb[5]        - y * aa[5];
    tr[j] = y;
  }
  /* backward pass, write trimmed + re-reversed result */
  float first = tr[TEXT - 1];
  for (int i = 0; i < 5; ++i) z[i] = zv[i] * first;
  for (int j = 0; j < TEXT; ++j) {
    float xt = tr[TEXT - 1 - j];
    float y = bb[0] * xt + z[0];
    z[0] = xt * bb[1] + z[1] - y * aa[1];
    z[1] = xt * bb[2] + z[2] - y * aa[2];
    z[2] = xt * bb[3] + z[3] - y * aa[3];
    z[3] = xt * bb[4] + z[4] - y * aa[4];
    z[4] = xt * bb[5]        - y * aa[5];
    int t = (TEXT - 1 - PAD) - j;       /* 3857 - j */
    if (t >= 0 && t < TT) vout[t] = y;
  }
}

/* ================= 4) Hpre = V @ W1 + b1 (WMMA) ================= */
__global__ __launch_bounds__(32) void k_hidden(const float* __restrict__ V,
                                               const _Float16* __restrict__ W1h,
                                               const float* __restrict__ b1,
                                               float* __restrict__ Hpre) {
  int rbase = blockIdx.x * 16;   /* 130 tiles -> 2080 rows */
  int hbase = blockIdx.y * 16;   /* 16 tiles -> 256 cols  */
  int lane = threadIdx.x;
  int lo = lane & 15, hi = lane >> 4;
  const float* arow = V + (size_t)(rbase + lo) * TT;
  v8f acc = {};
  for (int kb = 0; kb < TT; kb += 32) {
    v16h av, bv;
#pragma unroll
    for (int e = 0; e < 16; ++e) {
      int k = kb + kof(e, hi);
      av[e] = (_Float16)arow[k];
      bv[e] = W1h[(size_t)k * HH + hbase + lo];
    }
    acc = __builtin_amdgcn_wmma_f32_16x16x32_f16(false, av, false, bv,
                                                 (short)0, acc, false, false);
  }
#pragma unroll
  for (int r = 0; r < 8; ++r) {
    int row = rbase + r + 8 * hi;
    int col = hbase + lo;
    Hpre[(size_t)row * HH + col] = acc[r] + b1[col];
  }
}

/* ================= 5) fnV = relu(Hpre) @ W2 + b2 ================= */
__global__ void k_fn(const float* __restrict__ Hpre, const float* __restrict__ W2,
                     const float* __restrict__ b2, float* __restrict__ fnV) {
  int idx = blockIdx.x * blockDim.x + threadIdx.x;
  if (idx >= NROW * MMO) return;
  int row = idx / MMO, m = idx % MMO;
  const float* hr = Hpre + (size_t)row * HH;
  float s = b2[m];
  for (int h = 0; h < HH; ++h) s += fmaxf(hr[h], 0.f) * W2[h * MMO + m];
  fnV[idx] = s;
}

/* ================= 6) Wout (first 10240 floats of d_out) ================= */
__global__ void k_wout(const float* __restrict__ fnV, float* __restrict__ out) {
  int idx = blockIdx.x * blockDim.x + threadIdx.x;
  if (idx >= BBAT * MMO) return;
  int b = idx / MMO, m = idx % MMO;
  float vals[FF];
  float s = 0.f;
  for (int fo = 0; fo < FF; ++fo) {
    int f = FF - 1 - fo;
    float wr = fnV[((size_t)f * BBAT + b) * MMO + m] -
               fnV[((size_t)(f + 1) * BBAT + b) * MMO + m];
    wr = fmaxf(wr, 0.f);
    wr = rintf(wr * 1000.f) * 0.001f;   /* round half-even, 3 decimals */
    vals[fo] = wr;
    s += wr;
  }
  float inv = 1.f / (s + 1e-8f);
  for (int fo = 0; fo < FF; ++fo)
    out[((size_t)b * MMO + m) * FF + fo] = vals[fo] * inv;
}

/* ========== 7) U = diag(c)*W2 packed in B-fragment layout (f16) ========== */
__global__ __launch_bounds__(32) void k_ufrag(const float* __restrict__ Hpre,
                                              const float* __restrict__ W2,
                                              _Float16* __restrict__ U) {
  int p = blockIdx.x;             /* (f,b) pair: rowS = p, rowE = p+32 */
  int lane = threadIdx.x;
  int n = lane & 15, hi = lane >> 4;
  const float* hs = Hpre + (size_t)p * HH;
  const float* he = hs + (size_t)BBAT * HH;
#pragma unroll
  for (int ks = 0; ks < 8; ++ks) {
    v16h v;
#pragma unroll
    for (int e = 0; e < 16; ++e) {
      int h = ks * 32 + kof(e, hi);
      float c = 0.5f * ((hs[h] > 0.f ? 1.f : 0.f) + (he[h] > 0.f ? 1.f : 0.f));
      v[e] = (_Float16)((n < MMO) ? c * W2[h * MMO + n] : 0.f);
    }
    ((v16h*)U)[(size_t)(p * 8 + ks) * 32 + lane] = v;
  }
}

/* ================= 8) SG = (W1 diag(c) W2) * diff (WMMA, hot) =================
   W1 t-strip (8 KB f16) staged into LDS by the Tensor Data Mover (wave 0),
   tracked with TENSORcnt; workgroup barrier publishes it to all 8 waves.   */
__global__ __launch_bounds__(256) void k_sg(const float* __restrict__ V,
                                            const _Float16* __restrict__ W1h,
                                            const _Float16* __restrict__ U,
                                            float* __restrict__ out) {
  __shared__ _Float16 W1s[16 * HH];   /* 8192 B strip of W1, f16 */
  int t0 = blockIdx.x * 16;

  if (threadIdx.x == 0) {
    /* 1-D TDM tile: 4096 contiguous f16 elements, global -> LDS */
    unsigned long long ga = (unsigned long long)(uintptr_t)(W1h + (size_t)t0 * HH);
    unsigned int lds = (unsigned int)(uintptr_t)(&W1s[0]);   /* flat[31:0] = LDS offset */
    v4u g0;
    g0[0] = 1u;                                   /* count=1, user descriptor     */
    g0[1] = lds;                                  /* lds_addr                     */
    g0[2] = (unsigned int)(ga & 0xFFFFFFFFu);     /* global_addr[31:0]            */
    g0[3] = (unsigned int)((ga >> 32) & 0x1FFFFFFu) | 0x80000000u; /* [56:32]+type=2 */
    v8i g1;
    g1[0] = 0x00010000;                           /* data_size=1 (2 bytes)        */
    g1[1] = (int)((4096u & 0xFFFFu) << 16);       /* tensor_dim0 lo16 @ [63:48]   */
    g1[2] = (int)((4096u >> 16) | (1u << 16));    /* tensor_dim0 hi16, dim1=1     */
    g1[3] = (int)(4096u << 16);                   /* tensor_dim1 hi=0, tile_dim0  */
    g1[4] = 1;                                    /* tile_dim1=1, tile_dim2=0     */
    g1[5] = 4096;                                 /* tensor_dim0_stride[31:0]     */
    g1[6] = 0;                                    /* stride hi, dim1_stride lo    */
    g1[7] = 0;
    v4i gz = {0, 0, 0, 0};
#if defined(__clang_major__) && (__clang_major__ >= 23)
    v8i gz8 = {0, 0, 0, 0, 0, 0, 0, 0};
    __builtin_amdgcn_tensor_load_to_lds(g0, g1, gz, gz, gz8, 0);
#else
    __builtin_amdgcn_tensor_load_to_lds(g0, g1, gz, gz, 0);
#endif
    __builtin_amdgcn_s_wait_tensorcnt(0);
  }
  __syncthreads();

  int tid = threadIdx.x;
  int lane = tid & 31, wave = tid >> 5;
  int lo = lane & 15, hi = lane >> 4;

  /* A fragments (W1 t-strip) cached in registers for all 16 (f,b) pairs */
  v16h af[8];
#pragma unroll
  for (int ks = 0; ks < 8; ++ks)
#pragma unroll
    for (int e = 0; e < 16; ++e)
      af[ks][e] = W1s[lo * HH + ks * 32 + kof(e, hi)];

  for (int i = 0; i < 16; ++i) {
    int p = blockIdx.y * 128 + wave * 16 + i;   /* 0..2047 */
    int f = p >> 5, b = p & 31;
    const v16h* Up = (const v16h*)U + (size_t)p * 8 * 32 + lane;
    __builtin_prefetch((const void*)(Up + 8 * 32), 0, 0);   /* global_prefetch */
    v8f acc = {};
#pragma unroll
    for (int ks = 0; ks < 8; ++ks) {
      v16h bf = Up[(size_t)ks * 32];
      acc = __builtin_amdgcn_wmma_f32_16x16x32_f16(false, af[ks], false, bf,
                                                   (short)0, acc, false, false);
    }
    if (lo < MMO) {   /* N columns 0..4 carry real data */
      int tb = t0 + 8 * hi;                             /* 8 consecutive t   */
      const float4* vs4 = (const float4*)(V + (size_t)p * TT + tb);
      const float4* ve4 = (const float4*)(V + (size_t)(p + BBAT) * TT + tb);
      float4 a0 = vs4[0], a1 = vs4[1];
      float4 e0 = ve4[0], e1 = ve4[1];
      float4 o0, o1;
      o0.x = acc[0] * (a0.x - e0.x);
      o0.y = acc[1] * (a0.y - e0.y);
      o0.z = acc[2] * (a0.z - e0.z);
      o0.w = acc[3] * (a0.w - e0.w);
      o1.x = acc[4] * (a1.x - e1.x);
      o1.y = acc[5] * (a1.y - e1.y);
      o1.z = acc[6] * (a1.z - e1.z);
      o1.w = acc[7] * (a1.w - e1.w);
      size_t ob = (size_t)BBAT * MMO * FF +             /* skip Wout (10240) */
                  ((size_t)(b * MMO + lo) * FF + (FF - 1 - f)) * TT;
      float4* op = (float4*)(out + ob + tb);
      op[0] = o0;
      op[1] = o1;
    }
  }
}

/* ================================================================== */
extern "C" void kernel_launch(void* const* d_in, const int* in_sizes, int n_in,
                              void* d_out, int out_size, void* d_ws, size_t ws_size,
                              hipStream_t stream) {
  const float* x  = (const float*)d_in[0];
  const float* W1 = (const float*)d_in[1];
  const float* b1 = (const float*)d_in[2];
  const float* W2 = (const float*)d_in[3];
  const float* b2 = (const float*)d_in[4];
  float* out = (float*)d_out;

  float* ws   = (float*)d_ws;
  float* coef = ws + OFF_COEF;
  float* V    = ws + OFF_V;
  float* tmp  = ws + OFF_TMP;
  float* Hpre = ws + OFF_HPRE;
  float* fnV  = ws + OFF_FN;
  _Float16* U   = (_Float16*)(ws + OFF_U);
  _Float16* W1h = (_Float16*)(ws + OFF_W1H);

  k_design<<<1, 64, 0, stream>>>(coef);
  k_copy_x<<<(BBAT * TT + 255) / 256, 256, 0, stream>>>(x, V);
  k_cvt<<<(TT * HH + 255) / 256, 256, 0, stream>>>(W1, W1h);
  k_filt<<<NFB / 64, 64, 0, stream>>>(x, coef, V, tmp);
  k_hidden<<<dim3(NROW / 16, HH / 16), 32, 0, stream>>>(V, W1h, b1, Hpre);
  k_fn<<<(NROW * MMO + 255) / 256, 256, 0, stream>>>(Hpre, W2, b2, fnV);
  k_wout<<<1, BBAT * MMO, 0, stream>>>(fnV, out);
  k_ufrag<<<NFB, 32, 0, stream>>>(Hpre, W2, U);
  k_sg<<<dim3(TT / 16, 16), 256, 0, stream>>>(V, W1h, U, out);
}
